// KeypointDetector_29523605193335
// MI455X (gfx1250) — matrix operand order repaired and checked by reading, not compiled, not run on hardware
//
#include <hip/hip_runtime.h>

// MI455X (gfx1250) KeypointDetector forward, WMMA f16 (f32 accumulate).
// One wave = one point's 16 neighbors (M-tile), 2 points/wave, 8 waves/block.
// All shared weights preconverted to f16 B-fragment layout in LDS once per
// block; B-frag fetch = 2x ds_load_b128, no per-use cvt.

typedef __attribute__((ext_vector_type(16))) _Float16 v16h;
typedef __attribute__((ext_vector_type(8)))  _Float16 v8h;
typedef __attribute__((ext_vector_type(8)))  float    v8f;
typedef __attribute__((ext_vector_type(4)))  float    v4f;
typedef __attribute__((ext_vector_type(2)))  float    v2f;

#define BNSC 0.9999995000003749f   // 1/sqrt(1 + 1e-6)

__device__ __forceinline__ v8f wmma_f16(v16h a, v16h b, v8f c) {
  return __builtin_amdgcn_wmma_f32_16x16x32_f16(false, a, false, b,
                                                (short)0, c, false, false);
}

__device__ __forceinline__ float lrelu(float x) {
  return fmaxf(x, 0.0f) + 0.2f * fminf(x, 0.0f);
}

// Preload W [Cout x Cin] (Cin multiple of 32) into LDS as f16 B-fragments.
// Fragment f = nt*ktiles + kt; per-lane 16 contiguous halves at f*512+lane*16.
// B layout used here (contiguous k, same as A): lanes 0-15 -> col = lane,   K = koff+0..15
//               lanes 16-31 -> col = lane-16, K = koff+16..31
__device__ __forceinline__ void preload_frags(const float* __restrict__ W,
                                              int Cin, int ntiles, int ktiles,
                                              _Float16* dst, int wid, int lane) {
  const int nf = ntiles * ktiles;
  for (int f = wid; f < nf; f += 8) {
    const int nt = f / ktiles, kt = f - nt * ktiles;
    const int n  = nt * 16 + (lane & 15);
    const int c0 = kt * 32 + ((lane & 16) ? 16 : 0);
    const float* p = W + n * Cin + c0;
    _Float16* q = dst + f * 512 + lane * 16;
#pragma unroll
    for (int i = 0; i < 16; i += 4) {
      v4f v = *(const v4f*)(p + i);
      q[i + 0] = (_Float16)v.x;
      q[i + 1] = (_Float16)v.y;
      q[i + 2] = (_Float16)v.z;
      q[i + 3] = (_Float16)v.w;
    }
  }
}

// Fetch one preloaded B-fragment: two ds_load_b128.
__device__ __forceinline__ v16h ld_bfrag(const _Float16* base, int f, int lane) {
  const v8h* p = (const v8h*)(base + f * 512 + lane * 16);
  v8h lo = p[0];
  v8h up = p[1];
  v16h b;
#pragma unroll
  for (int i = 0; i < 8; ++i) { b[i] = lo[i]; b[i + 8] = up[i]; }
  return b;
}

// A-frag (16x32 f16) from a row-major f16 LDS tile.
// A layout: lanes 0-15: M=lane, halves0-15=K+0..15
//           lanes 16-31: M=lane-16, halves0-15=K+16..31
__device__ __forceinline__ v16h load_afrag(const _Float16* S, int pitch,
                                           int kt, int lane) {
  const int m  = lane & 15;
  const int c0 = kt * 32 + ((lane & 16) ? 16 : 0);
  const _Float16* p = S + m * pitch + c0;
  v8h lo = *(const v8h*)(p);
  v8h up = *(const v8h*)(p + 8);
  v16h a;
#pragma unroll
  for (int i = 0; i < 8; ++i) { a[i] = lo[i]; a[i + 8] = up[i]; }
  return a;
}

// ---- LDS byte offsets ----
#define OFF_WFRAG   0         // 54 frags * 1024 B  (c1:0-1, c2:2-5, c3:6-21, wa:22-53)
#define OFF_STAGE   55296     // 8 waves * 16x128 f16 (reused for wp frags in head)
#define OFF_POOLED  88064     // 16 x 128 f16
#define OFF_HBUF32  92160     // 16 x 128 f32
#define OFF_HBUF16  100352    // 16 x 128 f16
#define OFF_MBUF    104448    // 16 x 64  f16
#define OFF_H2BUF   106496    // 16 x 32  f32
#define OFF_SCBI    108544    // 352 x float2 (BN scale/bias: c1|c2|c3|p)
#define LDS_BYTES   111360

__global__ __launch_bounds__(256)
void keypoint_fused_kernel(
    const float* __restrict__ pc,  const float* __restrict__ knn,
    const float* __restrict__ w1,  const float* __restrict__ b1,
    const float* __restrict__ g1,  const float* __restrict__ be1,
    const float* __restrict__ w2,  const float* __restrict__ b2,
    const float* __restrict__ g2,  const float* __restrict__ be2,
    const float* __restrict__ w3,  const float* __restrict__ b3,
    const float* __restrict__ g3,  const float* __restrict__ be3,
    const float* __restrict__ wa,
    const float* __restrict__ wp,  const float* __restrict__ bp,
    const float* __restrict__ gp,  const float* __restrict__ bep,
    const float* __restrict__ wm1, const float* __restrict__ bm1,
    const float* __restrict__ wm2, const float* __restrict__ bm2,
    const float* __restrict__ wm3, const float* __restrict__ bm3,
    float* __restrict__ out, int n_points, long long scalar_idx)
{
  __shared__ __align__(16) unsigned char smem[LDS_BYTES];
  _Float16* wfrag    = (_Float16*)(smem + OFF_WFRAG);
  _Float16* stageAll = (_Float16*)(smem + OFF_STAGE);
  _Float16* pooledB  = (_Float16*)(smem + OFF_POOLED);
  float*    hbuf32   = (float*)   (smem + OFF_HBUF32);
  _Float16* hbuf16   = (_Float16*)(smem + OFF_HBUF16);
  _Float16* mbuf     = (_Float16*)(smem + OFF_MBUF);
  float*    h2buf    = (float*)   (smem + OFF_H2BUF);
  v2f*      scbi     = (v2f*)     (smem + OFF_SCBI);

  _Float16* fragC1 = wfrag;               // 2 frags
  _Float16* fragC2 = wfrag + 2 * 512;     // 4 frags
  _Float16* fragC3 = wfrag + 6 * 512;     // 16 frags (nt*2+kt)
  _Float16* fragWA = wfrag + 22 * 512;    // 32 frags (nt*4+kt)

  const int tid   = threadIdx.x;
  const int lane  = tid & 31;
  const int wid   = tid >> 5;             // 0..7
  const bool hiL  = (lane & 16) != 0;
  const int mbase = hiL ? 8 : 0;          // C-tile row base for this lane
  const int col   = lane & 15;
  _Float16* stage = stageAll + wid * 2048;

  // ---------------- one-time per-block preload ----------------
  if (wid < 2) {  // conv1 B-frags, Cin=10 zero-padded to 32
    const int n  = wid * 16 + col;
    const int c0 = hiL ? 16 : 0;
    _Float16* q = fragC1 + wid * 512 + lane * 16;
#pragma unroll
    for (int i = 0; i < 16; ++i) {
      int c = c0 + i;
      q[i] = (c < 10) ? (_Float16)w1[n * 10 + c] : (_Float16)0.0f;
    }
  }
  preload_frags(w2, 32, 4, 1, fragC2, wid, lane);
  preload_frags(w3, 64, 8, 2, fragC3, wid, lane);
  preload_frags(wa, 128, 8, 4, fragWA, wid, lane);
  if (tid < 352) {  // folded BN (sc, bi) per channel: c1 | c2 | c3 | p
    const float *g, *b, *be; int ch;
    if      (tid < 32)  { g = g1; b = b1; be = be1; ch = tid; }
    else if (tid < 96)  { g = g2; b = b2; be = be2; ch = tid - 32; }
    else if (tid < 224) { g = g3; b = b3; be = be3; ch = tid - 96; }
    else                { g = gp; b = bp; be = bep; ch = tid - 224; }
    float sc = g[ch] * BNSC;
    v2f t; t.x = sc; t.y = b[ch] * sc + be[ch];
    scbi[tid] = t;
  }
  __syncthreads();

  // ---------------- per-point pipeline: 2 points per wave ----------------
  for (int p = 0; p < 2; ++p) {
    const int       prow = wid * 2 + p;                        // 0..15
    const long long gpt  = (long long)blockIdx.x * 16 + prow;  // flat (b,n)
    const float* P  = pc  + gpt * 3;
    const float* Kp = knn + gpt * 48 + (long long)col * 3;
    float px = P[0], py = P[1], pz = P[2];
    float kx = Kp[0], ky = Kp[1], kz = Kp[2];
    float rx = px - kx, ry = py - ky, rz = pz - kz;
    float dd = sqrtf(rx * rx + ry * ry + rz * rz);

    // feat row (10 ch zero-padded to 32): [dist, rel, point, knn]
    v16h a0 = {};
    if (!hiL) {
      a0[0] = (_Float16)dd; a0[1] = (_Float16)rx; a0[2] = (_Float16)ry;
      a0[3] = (_Float16)rz; a0[4] = (_Float16)px; a0[5] = (_Float16)py;
      a0[6] = (_Float16)pz; a0[7] = (_Float16)kx;
      a0[8] = (_Float16)ky; a0[9] = (_Float16)kz;
    }

    // ---- conv1: 10 -> 32 ----
#pragma unroll
    for (int nt = 0; nt < 2; ++nt) {
      v8f acc = {};
      acc = wmma_f16(a0, ld_bfrag(fragC1, nt, lane), acc);
      const int ch = nt * 16 + col;
      v2f sb = scbi[ch];
#pragma unroll
      for (int r = 0; r < 8; ++r)
        stage[(mbase + r) * 128 + ch] = (_Float16)lrelu(fmaf(acc[r], sb.x, sb.y));
    }

    // ---- conv2: 32 -> 64 ----
    {
      v16h a1 = load_afrag(stage, 128, 0, lane);
#pragma unroll
      for (int nt = 0; nt < 4; ++nt) {
        v8f acc = {};
        acc = wmma_f16(a1, ld_bfrag(fragC2, nt, lane), acc);
        const int ch = nt * 16 + col;
        v2f sb = scbi[32 + ch];
#pragma unroll
        for (int r = 0; r < 8; ++r)
          stage[(mbase + r) * 128 + ch] = (_Float16)lrelu(fmaf(acc[r], sb.x, sb.y));
      }
    }

    // ---- conv3: 64 -> 128 (keep activations in regs for pooling) ----
    v8f x3[8];
    {
      v16h a2k0 = load_afrag(stage, 128, 0, lane);
      v16h a2k1 = load_afrag(stage, 128, 1, lane);
#pragma unroll
      for (int nt = 0; nt < 8; ++nt) {
        v8f acc = {};
        acc = wmma_f16(a2k0, ld_bfrag(fragC3, nt * 2 + 0, lane), acc);
        acc = wmma_f16(a2k1, ld_bfrag(fragC3, nt * 2 + 1, lane), acc);
        const int ch = nt * 16 + col;
        v2f sb = scbi[96 + ch];
        v8f o;
#pragma unroll
        for (int r = 0; r < 8; ++r) {
          o[r] = lrelu(fmaf(acc[r], sb.x, sb.y));
          stage[(mbase + r) * 128 + ch] = (_Float16)o[r];
        }
        x3[nt] = o;
      }
    }

    // ---- attentive pooling: softmax over the 16 neighbor rows ----
    {
      v16h xa[4];
#pragma unroll
      for (int kt = 0; kt < 4; ++kt) xa[kt] = load_afrag(stage, 128, kt, lane);
#pragma unroll
      for (int nt = 0; nt < 8; ++nt) {
        v8f acc = {};
#pragma unroll
        for (int kt = 0; kt < 4; ++kt)
          acc = wmma_f16(xa[kt], ld_bfrag(fragWA, nt * 4 + kt, lane), acc);
        float mx = acc[0];
#pragma unroll
        for (int r = 1; r < 8; ++r) mx = fmaxf(mx, acc[r]);
        mx = fmaxf(mx, __shfl_xor(mx, 16, 32));
        float e[8], s = 0.0f;
#pragma unroll
        for (int r = 0; r < 8; ++r) { e[r] = __expf(acc[r] - mx); s += e[r]; }
        s += __shfl_xor(s, 16, 32);
        const float inv = 1.0f / s;
        float pl = 0.0f;
#pragma unroll
        for (int r = 0; r < 8; ++r) pl += x3[nt][r] * e[r];
        pl *= inv;
        pl += __shfl_xor(pl, 16, 32);
        if (!hiL) pooledB[prow * 128 + nt * 16 + col] = (_Float16)pl;
      }
    }
  }

  // stage region is now free: preload wp B-frags (32 frags) into it.
  preload_frags(wp, 128, 8, 4, stageAll, wid, lane);
  __syncthreads();

  // ---- wp: pooled[16x128] @ wp^T (+BN+lrelu) -> N128; one N-tile per wave ----
  {
    v16h pa[4];
#pragma unroll
    for (int kt = 0; kt < 4; ++kt) pa[kt] = load_afrag(pooledB, 128, kt, lane);
    v8f acc = {};
#pragma unroll
    for (int kt = 0; kt < 4; ++kt)
      acc = wmma_f16(pa[kt], ld_bfrag(stageAll, wid * 4 + kt, lane), acc);
    const int ch = wid * 16 + col;
    v2f sb = scbi[224 + ch];
#pragma unroll
    for (int r = 0; r < 8; ++r) {
      float v = lrelu(fmaf(acc[r], sb.x, sb.y));
      hbuf32[(mbase + r) * 128 + ch] = v;
      hbuf16[(mbase + r) * 128 + ch] = (_Float16)v;
    }
  }
  __syncthreads();

  // ---- coalesced N128 output: 16 points x 128 f32 as b128 stores ----
  {
    const long long base = (long long)n_points + (long long)blockIdx.x * 2048;
    const v4f* src = (const v4f*)hbuf32;
    v4f* dst = (v4f*)(out + base);
    dst[tid]       = src[tid];
    dst[tid + 256] = src[tid + 256];
  }

  // ---- MLP head (wave 0): 128 -> 64 -> 32 -> 1, relu ----
  if (wid == 0) {
    v16h ha[4];
#pragma unroll
    for (int kt = 0; kt < 4; ++kt) ha[kt] = load_afrag(hbuf16, 128, kt, lane);
#pragma unroll
    for (int nt = 0; nt < 4; ++nt) {
      v8f acc = {};
#pragma unroll
      for (int kt = 0; kt < 4; ++kt) {
        // wm1 B-frag from global (head runs once per block)
        const int n  = nt * 16 + col;
        const int c0 = kt * 32 + (hiL ? 16 : 0);
        const v4f* wv = (const v4f*)(wm1 + n * 128 + c0);
        v16h bf;
#pragma unroll
        for (int i = 0; i < 4; ++i) {
          v4f q = wv[i];
          bf[4 * i + 0] = (_Float16)q.x; bf[4 * i + 1] = (_Float16)q.y;
          bf[4 * i + 2] = (_Float16)q.z; bf[4 * i + 3] = (_Float16)q.w;
        }
        acc = wmma_f16(ha[kt], bf, acc);
      }
      const int ch = nt * 16 + col;
      const float bi = bm1[ch];
#pragma unroll
      for (int r = 0; r < 8; ++r)
        mbuf[(mbase + r) * 64 + ch] = (_Float16)(acc[r] + bi);
    }
    v16h ma0 = load_afrag(mbuf, 64, 0, lane);
    v16h ma1 = load_afrag(mbuf, 64, 1, lane);
#pragma unroll
    for (int nt = 0; nt < 2; ++nt) {
      v8f acc = {};
#pragma unroll
      for (int kt = 0; kt < 2; ++kt) {
        const int n  = nt * 16 + col;
        const int c0 = kt * 32 + (hiL ? 16 : 0);
        const v4f* wv = (const v4f*)(wm2 + n * 64 + c0);
        v16h bf;
#pragma unroll
        for (int i = 0; i < 4; ++i) {
          v4f q = wv[i];
          bf[4 * i + 0] = (_Float16)q.x; bf[4 * i + 1] = (_Float16)q.y;
          bf[4 * i + 2] = (_Float16)q.z; bf[4 * i + 3] = (_Float16)q.w;
        }
        acc = wmma_f16(kt == 0 ? ma0 : ma1, bf, acc);
      }
      const int ch = nt * 16 + col;
      const float bi = bm2[ch];
#pragma unroll
      for (int r = 0; r < 8; ++r)
        h2buf[(mbase + r) * 32 + ch] = acc[r] + bi;
    }
    if (lane < 16) {  // final 32 -> 1 dot + relu -> score
      float a = bm3[0];
      const float* row = h2buf + lane * 32;
#pragma unroll
      for (int c = 0; c < 32; ++c) a += row[c] * wm3[c];
      out[(long long)blockIdx.x * 16 + lane] = fmaxf(a, 0.0f);
    }
  }

  if (blockIdx.x == 0 && tid == 0) out[scalar_idx] = 0.0f;  // trailing scalar
}

extern "C" void kernel_launch(void* const* d_in, const int* in_sizes, int n_in,
                              void* d_out, int out_size, void* d_ws, size_t ws_size,
                              hipStream_t stream) {
  const float* pc  = (const float*)d_in[0];
  const float* kn  = (const float*)d_in[1];
  const float* w1  = (const float*)d_in[2];
  const float* b1  = (const float*)d_in[3];
  const float* g1  = (const float*)d_in[4];
  const float* be1 = (const float*)d_in[5];
  const float* w2  = (const float*)d_in[6];
  const float* b2  = (const float*)d_in[7];
  const float* g2  = (const float*)d_in[8];
  const float* be2 = (const float*)d_in[9];
  const float* w3  = (const float*)d_in[10];
  const float* b3  = (const float*)d_in[11];
  const float* g3  = (const float*)d_in[12];
  const float* be3 = (const float*)d_in[13];
  const float* wa  = (const float*)d_in[14];
  const float* wp  = (const float*)d_in[15];
  const float* bp  = (const float*)d_in[16];
  const float* gp  = (const float*)d_in[17];
  const float* bep = (const float*)d_in[18];
  const float* wm1 = (const float*)d_in[19];
  const float* bm1 = (const float*)d_in[20];
  const float* wm2 = (const float*)d_in[21];
  const float* bm2 = (const float*)d_in[22];
  const float* wm3 = (const float*)d_in[23];
  const float* bm3 = (const float*)d_in[24];

  const int n_points = in_sizes[0] / 3;     // B*N = 65536
  const int blocks   = n_points / 16;       // 16 points per block

  keypoint_fused_kernel<<<blocks, 256, 0, stream>>>(
      pc, kn, w1, b1, g1, be1, w2, b2, g2, be2, w3, b3, g3, be3,
      wa, wp, bp, gp, bep, wm1, bm1, wm2, bm2, wm3, bm3,
      (float*)d_out, n_points, (long long)out_size - 1);
}
